// Shapelet_75634374082895
// MI455X (gfx1250) — compile-verified, matrix operand order checked by reading, not run on hardware
//
#include <hip/hip_runtime.h>
#include <hip/hip_bf16.h>
#include <math.h>

typedef __attribute__((ext_vector_type(16))) _Float16 v16h;
typedef __attribute__((ext_vector_type(8)))  float    v8f;
typedef __attribute__((ext_vector_type(4)))  float    v4f;

#define TAUF     10.0f
#define T_TOTAL  2048
#define T_OUT    2017          // (2048-32)/1 + 1
#define LWIN     32
#define NSHAPE   32
#define MCH      8
#define BATCH    8
#define NTILES   127           // ceil(2017/16)
#define T_PAD    (NTILES * 16) // 2032
#define WAVES    8
#define BIGVAL   1.0e30f

// One workgroup per (b,m). 8 wave32s stride the 127 t-tiles.
// dot[t,n] = sum_l x[b,m,t+l]*w[n,m,l]  via one v_wmma_f32_16x16x32_f16
// per 16(t) x 16(n) tile (K = 32 = window length, exact match).
__global__ __launch_bounds__(256) void shapelet_wmma_kernel(
    const float* __restrict__ x,
    const float* __restrict__ w,
    float* __restrict__ out)
{
    __shared__ float sX2[T_PAD];            // ||x_win||^2 per window (f32 exact)
    __shared__ float sM[WAVES][NSHAPE];
    __shared__ float sS[WAVES][NSHAPE];
    __shared__ float sD[WAVES][NSHAPE];

    const int bm   = blockIdx.x;
    const int b    = bm >> 3;
    const int m    = bm & 7;
    const int tid  = threadIdx.x;
    const int wave = tid >> 5;
    const int lane = tid & 31;
    const int half = lane >> 4;     // which 16-lane half
    const int lmod = lane & 15;

    const float* xrow = x + (size_t)(b * MCH + m) * T_TOTAL;

    // ---- Phase 0: all window norms into LDS; pad windows get BIGVAL so the
    //      tail is masked arithmetically (no branches in the hot loop). ----
    for (int tt = tid; tt < T_PAD; tt += 256) {
        float s = BIGVAL;
        if (tt < T_OUT) {
            s = 0.0f;
            #pragma unroll
            for (int l = 0; l < LWIN; ++l) {
                const float v = xrow[tt + l];
                s = fmaf(v, v, s);
            }
        }
        sX2[tt] = s;
    }

    // ---- Load B fragments (two n-tiles) once; exact f32 ||w||^2 ----
    // 16-bit A/B fragment K pattern per lane: e=0..7 -> k = 8*half + e,
    // e=8..15 -> k = 16 + 8*half + (e-8).  (ISA 7.12.2 16-bit layouts)
    v16h bfrag[2];
    float w2[2];
    #pragma unroll
    for (int nt = 0; nt < 2; ++nt) {
        const int n = lmod + 16 * nt;
        const float* wrow = w + (size_t)(n * MCH + m) * LWIN;
        float s = 0.0f;
        #pragma unroll
        for (int e = 0; e < 16; ++e) {
            const int k = (e & 7) + 8 * half + ((e >> 3) << 4);
            const float v = wrow[k];
            s = fmaf(v, v, s);
            bfrag[nt][e] = (_Float16)v;
        }
        s += __shfl_xor(s, 16, 32);   // partner lane holds the other 16 l's
        w2[nt] = s;
    }
    __syncthreads();

    // ---- Streaming soft-min (logsumexp) + hard-min state, per n-tile ----
    float Mx[2]   = { -BIGVAL, -BIGVAL };  // running max of -TAU*clip(d2)
    float Sx[2]   = { 0.0f, 0.0f };        // running sum of exp(v - Mx)
    float dmin[2] = { BIGVAL, BIGVAL };    // running min of clip(d2)

    for (int tile = wave; tile < NTILES; tile += WAVES) {
        const int t0   = tile * 16;
        const int trow = t0 + lmod;                       // A-fragment row M
        const int trd  = trow > (T_OUT - 1) ? (T_OUT - 1) : trow; // clamp OOB
        const float* xp = xrow + trd;

        v16h a;
        #pragma unroll
        for (int e = 0; e < 16; ++e) {
            const int k = (e & 7) + 8 * half + ((e >> 3) << 4);
            a[e] = (_Float16)xp[k];
        }

        // ||x_win||^2 for this lane's 8 output rows: consecutive t values at a
        // 32B-aligned LDS offset -> two ds_load_b128, broadcast across lanes.
        const v4f* px2 = (const v4f*)&sX2[t0 + 8 * half];
        const v4f xa = px2[0];
        const v4f xb = px2[1];
        float x2v[8];
        #pragma unroll
        for (int r = 0; r < 4; ++r) { x2v[r] = xa[r]; x2v[r + 4] = xb[r]; }

        #pragma unroll
        for (int nt = 0; nt < 2; ++nt) {
            v8f c = {};
            c = __builtin_amdgcn_wmma_f32_16x16x32_f16(
                    /*neg_a=*/false, a, /*neg_b=*/false, bfrag[nt],
                    /*c_mod=*/(short)0, c, /*reuse_a=*/false, /*reuse_b=*/false);
            #pragma unroll
            for (int r = 0; r < 8; ++r) {
                float d2 = fmaf(-2.0f, c[r], x2v[r] + w2[nt]);
                d2 = fmaxf(d2, 0.0f);                 // clip
                dmin[nt] = fminf(dmin[nt], d2);
                const float vI = -TAUF * d2;          // TAU * logits_t
                const float Mn = fmaxf(Mx[nt], vI);   // branch-free online LSE
                Sx[nt] = fmaf(Sx[nt], __expf(Mx[nt] - Mn), __expf(vI - Mn));
                Mx[nt] = Mn;
            }
        }
    }

    // ---- Merge lane <-> lane+16 (same n, complementary t rows) ----
    #pragma unroll
    for (int nt = 0; nt < 2; ++nt) {
        const float Mo  = __shfl_xor(Mx[nt],   16, 32);
        const float So  = __shfl_xor(Sx[nt],   16, 32);
        const float dmo = __shfl_xor(dmin[nt], 16, 32);
        const float Mn  = fmaxf(Mx[nt], Mo);
        Sx[nt]   = fmaf(Sx[nt], __expf(Mx[nt] - Mn), So * __expf(Mo - Mn));
        Mx[nt]   = Mn;
        dmin[nt] = fminf(dmin[nt], dmo);
    }

    // ---- Cross-wave reduction in LDS ----
    if (lane < 16) {
        #pragma unroll
        for (int nt = 0; nt < 2; ++nt) {
            sM[wave][lmod + 16 * nt] = Mx[nt];
            sS[wave][lmod + 16 * nt] = Sx[nt];
            sD[wave][lmod + 16 * nt] = dmin[nt];
        }
    }
    __syncthreads();

    if (tid < NSHAPE) {
        const int n = tid;
        float M  = sM[0][n];
        float S  = sS[0][n];
        float dm = sD[0][n];
        #pragma unroll
        for (int wv = 1; wv < WAVES; ++wv) {
            const float Mo = sM[wv][n];
            const float So = sS[wv][n];
            const float Mn = fmaxf(M, Mo);
            S = fmaf(S, __expf(M - Mn), So * __expf(Mo - Mn));
            M = Mn;
            dm = fminf(dm, sD[wv][n]);
        }
        const float pooled = (M + __logf(S)) / TAUF;
        const float pred   = __expf(pooled);
        const float dmn    = sqrtf(dm);   // dm already clipped >= 0
        const int base = (b * NSHAPE + n) * MCH + m;   // (b, n, m) flat
        out[base] = pred;
        out[BATCH * NSHAPE * MCH + base] = dmn;
    }
}

extern "C" void kernel_launch(void* const* d_in, const int* in_sizes, int n_in,
                              void* d_out, int out_size, void* d_ws, size_t ws_size,
                              hipStream_t stream) {
    const float* x   = (const float*)d_in[0];   // (8, 8, 2048) f32
    const float* wts = (const float*)d_in[1];   // (32, 8, 32) f32
    float* out = (float*)d_out;                 // 2048 pred + 2048 d_min
    (void)in_sizes; (void)n_in; (void)out_size; (void)d_ws; (void)ws_size;
    shapelet_wmma_kernel<<<BATCH * MCH, 256, 0, stream>>>(x, wts, out);
}
